// BahdanauAttention_41472204210353
// MI455X (gfx1250) — matrix operand order repaired and checked by reading, compiled, not run on hardware
//
#include <hip/hip_runtime.h>
#include <math.h>

// Problem constants (match reference): B=64, S=2048, D=512, UNITS=256
#define B_DIM 64
#define S_DIM 2048
#define D_DIM 512
#define U_DIM 256
#define NS    16            // S-chunks for context partial sums (2048/16 = 128 s per chunk)
#define SROWS 64            // s-rows per score block (4 WMMA sub-tiles share each B fragment)

// CDNA5 WMMA vector types (probe-verified signatures)
typedef __attribute__((ext_vector_type(16))) __bf16    v16bf;
typedef __attribute__((ext_vector_type(8)))  float     v8f;
typedef __attribute__((ext_vector_type(4)))  unsigned  v4u;

// f32 -> bf16 with round-to-nearest-even (bit pattern; no __bf16 arithmetic needed)
static __device__ __forceinline__ unsigned short f32_to_bf16(float f) {
    union { float f; unsigned u; } v; v.f = f;
    unsigned u = v.u;
    unsigned rnd = 0x7FFFu + ((u >> 16) & 1u);
    return (unsigned short)((u + rnd) >> 16);
}

// ---------------------------------------------------------------------------
// Kernel 1: W1 [D][U] f32  ->  W1^T [U][D] bf16 (workspace)
// Transposed so a WMMA B-fragment (16 consecutive K for one column u) is a
// contiguous 32-byte global load.
// ---------------------------------------------------------------------------
__global__ __launch_bounds__(256) void w1t_kernel(const float* __restrict__ W1,
                                                  unsigned short* __restrict__ w1t) {
    const int d = blockIdx.x;      // 0..511
    const int u = threadIdx.x;     // 0..255 (coalesced read of W1 row)
    w1t[(size_t)u * D_DIM + d] = f32_to_bf16(W1[(size_t)d * U_DIM + u]);
}

// ---------------------------------------------------------------------------
// Kernel 2: pqc[b][u] = query[b]·W2[:,u] + b2[u] + b1[u]   (b1 folded in)
// 16.8 MFLOP total -> plain f32, coalesced over u.
// ---------------------------------------------------------------------------
__global__ __launch_bounds__(256) void pq_kernel(const float* __restrict__ query,
                                                 const float* __restrict__ W2,
                                                 const float* __restrict__ b2,
                                                 const float* __restrict__ b1,
                                                 float* __restrict__ pqc) {
    const int b = blockIdx.x;
    const int u = threadIdx.x;
    const float* q = query + (size_t)b * D_DIM;
    float acc = 0.0f;
    for (int d = 0; d < D_DIM; ++d)
        acc += q[d] * W2[(size_t)d * U_DIM + u];
    pqc[(size_t)b * U_DIM + u] = acc + b2[u] + b1[u];
}

// ---------------------------------------------------------------------------
// Kernel 3 (the WMMA kernel): fused  score[b,s] = tanh(values@W1 + b1 + pq)·V + bV
// Workgroup: 512 threads = 16 waves. Each block owns (b, 64-row s-tile).
// Wave w owns u-columns [16w, 16w+16). Per K-step it loads ONE B fragment and
// reuses it from registers across FOUR 16-row A sub-tiles (4 accumulators),
// cutting W1^T L2 traffic 4x vs. a 16-row block. pv is never materialized.
// ---------------------------------------------------------------------------
__global__ __launch_bounds__(512) void score_kernel(const float* __restrict__ values,
                                                    const unsigned short* __restrict__ w1t,
                                                    const float* __restrict__ pqc,
                                                    const float* __restrict__ Vvec,
                                                    const float* __restrict__ bV,
                                                    float* __restrict__ score) {
    // A tile staged as bf16 in LDS; row pad of 8 halves (16B) staggers banks so
    // the 16 lanes' ds_load_b128 of a K-slab hit distinct bank groups.
    __shared__ unsigned short aTile[SROWS][D_DIM + 8];   // ~65 KB
    __shared__ float wSum[16][SROWS];                    // [wave][row] partials

    const int b   = blockIdx.x >> 5;          // S/64 = 32 tiles per batch
    const int s0  = (blockIdx.x & 31) << 6;   // tile * 64
    const int tid = threadIdx.x;

    // Cooperative stage: 64 x 512 f32 -> bf16 into LDS (64 elems/thread)
    const float* vbase = values + ((size_t)b * S_DIM + s0) * D_DIM;
    #pragma unroll 4
    for (int i = tid; i < SROWS * D_DIM; i += 512) {
        const int r = i >> 9;          // row 0..63
        const int c = i & (D_DIM - 1); // col 0..511
        aTile[r][c] = f32_to_bf16(vbase[(size_t)r * D_DIM + c]);
    }
    __syncthreads();

    const int wave    = tid >> 5;          // 0..15 -> u-tile
    const int lane    = tid & 31;
    const int l16     = lane & 15;
    const int halfsel = lane >> 4;         // 0: lanes 0-15, 1: lanes 16-31
    const int u0      = wave << 4;

    // fragment addressing used by this code (wave32, contiguous K, same order for A and B):
    //   A (16x32 bf16): M = lane&15; lanes<16 hold K = k0+[0..16), lanes>=16 K = k0+16+[0..16)
    //   B (32x16 bf16): N = lane&15; lanes<16 hold K = k0+[0..16), lanes>=16 K = k0+16+[0..16)
    const int klo = halfsel << 4;                                   // 0 or 16
    const unsigned short* brow = w1t + (size_t)(u0 + l16) * D_DIM + (halfsel << 4);

    v8f acc0 = {}, acc1 = {}, acc2 = {}, acc3 = {};

    #pragma unroll
    for (int k0 = 0; k0 < D_DIM; k0 += 32) {
        // B fragment: 32 contiguous bytes of W1^T row (16 consecutive K);
        // loaded once, reused by 4 WMMAs below.
        v16bf bfrag;
        {
            const unsigned short* bp = brow + k0;
            v4u* bq = reinterpret_cast<v4u*>(&bfrag);
            bq[0] = *reinterpret_cast<const v4u*>(bp);
            bq[1] = *reinterpret_cast<const v4u*>(bp + 8);
        }
        // Four A sub-tiles (s-rows t*16 .. t*16+15), each two 16B LDS loads.
        #pragma unroll
        for (int t = 0; t < 4; ++t) {
            v16bf afrag;
            const unsigned short* alo = &aTile[(t << 4) + l16][k0 + klo];
            v4u* ap = reinterpret_cast<v4u*>(&afrag);
            ap[0] = *reinterpret_cast<const v4u*>(alo);
            ap[1] = *reinterpret_cast<const v4u*>(alo + 8);
            v8f& acc = (t == 0) ? acc0 : (t == 1) ? acc1 : (t == 2) ? acc2 : acc3;
            acc = __builtin_amdgcn_wmma_f32_16x16x32_bf16(
                      false, afrag, false, bfrag, (short)0, acc, false, false);
        }
    }

    // Epilogue: t = tanh(pv + pq') * V[u]; reduce over the 16 u's of this wave.
    // C layout: reg r, lane L -> row M = r + 8*(L>=16), col N = L&15.
    const int   u   = u0 + l16;
    const float pqu = pqc[(size_t)b * U_DIM + u];
    const float vu  = Vvec[u];

    #pragma unroll
    for (int t = 0; t < 4; ++t) {
        const v8f& acc = (t == 0) ? acc0 : (t == 1) ? acc1 : (t == 2) ? acc2 : acc3;
        #pragma unroll
        for (int r = 0; r < 8; ++r) {
            float x = tanhf(acc[r] + pqu) * vu;
            // butterfly sum across the 16 lanes holding the 16 u-columns
            x += __shfl_xor(x, 8, 16);
            x += __shfl_xor(x, 4, 16);
            x += __shfl_xor(x, 2, 16);
            x += __shfl_xor(x, 1, 16);
            if (l16 == 0) wSum[wave][(t << 4) + r + (halfsel << 3)] = x;
        }
    }
    __syncthreads();

    // Deterministic fixed-order reduction over the 16 waves (no FP atomics).
    if (tid < SROWS) {
        float s = 0.0f;
        #pragma unroll
        for (int w = 0; w < 16; ++w) s += wSum[w][tid];
        score[(size_t)b * S_DIM + s0 + tid] = s + bV[0];
    }
}

// ---------------------------------------------------------------------------
// Kernel 4: softmax over S per batch; writes attn straight into d_out slot.
// ---------------------------------------------------------------------------
__global__ __launch_bounds__(256) void softmax_kernel(const float* __restrict__ score,
                                                      float* __restrict__ attn) {
    __shared__ float red[256];
    const int b   = blockIdx.x;
    const int tid = threadIdx.x;
    const float* s = score + (size_t)b * S_DIM;

    float v[8];
    float mx = -INFINITY;
    #pragma unroll
    for (int i = 0; i < 8; ++i) { v[i] = s[tid + (i << 8)]; mx = fmaxf(mx, v[i]); }

    red[tid] = mx; __syncthreads();
    for (int off = 128; off > 0; off >>= 1) {
        if (tid < off) red[tid] = fmaxf(red[tid], red[tid + off]);
        __syncthreads();
    }
    mx = red[0]; __syncthreads();

    float sum = 0.0f;
    #pragma unroll
    for (int i = 0; i < 8; ++i) { v[i] = expf(v[i] - mx); sum += v[i]; }

    red[tid] = sum; __syncthreads();
    for (int off = 128; off > 0; off >>= 1) {
        if (tid < off) red[tid] += red[tid + off];
        __syncthreads();
    }
    const float inv = 1.0f / red[0];

    float* a = attn + (size_t)b * S_DIM;
    #pragma unroll
    for (int i = 0; i < 8; ++i) a[tid + (i << 8)] = v[i] * inv;
}

// ---------------------------------------------------------------------------
// Kernel 5: context partials. Grid (B*NS); each block accumulates 128 s-rows
// for all 512 d (2 per thread, coalesced). Streams `values` exactly once.
// ---------------------------------------------------------------------------
__global__ __launch_bounds__(256) void ctx_partial_kernel(const float* __restrict__ values,
                                                          const float* __restrict__ attn,
                                                          float* __restrict__ partial) {
    const int b     = blockIdx.x >> 4;
    const int chunk = blockIdx.x & (NS - 1);
    const int tid   = threadIdx.x;
    const int sBeg  = chunk * (S_DIM / NS);

    const float* vb = values + ((size_t)b * S_DIM + sBeg) * D_DIM;
    const float* ab = attn + (size_t)b * S_DIM + sBeg;

    float a0 = 0.0f, a1 = 0.0f;
    for (int s = 0; s < S_DIM / NS; ++s) {
        const float w = ab[s];                       // uniform -> scalarized
        const float* row = vb + (size_t)s * D_DIM;
        a0 += w * row[tid];
        a1 += w * row[tid + 256];
    }
    float* p = partial + ((size_t)(b * NS + chunk)) * D_DIM;
    p[tid]       = a0;
    p[tid + 256] = a1;
}

// Kernel 6: deterministic fixed-order reduce of the NS partials -> context.
__global__ __launch_bounds__(512) void ctx_reduce_kernel(const float* __restrict__ partial,
                                                         float* __restrict__ ctx) {
    const int b = blockIdx.x;
    const int d = threadIdx.x;
    float acc = 0.0f;
    #pragma unroll
    for (int c = 0; c < NS; ++c)
        acc += partial[((size_t)(b * NS + c)) * D_DIM + d];
    ctx[(size_t)b * D_DIM + d] = acc;
}

// ---------------------------------------------------------------------------
extern "C" void kernel_launch(void* const* d_in, const int* in_sizes, int n_in,
                              void* d_out, int out_size, void* d_ws, size_t ws_size,
                              hipStream_t stream) {
    // Inputs per setup_inputs() order
    const float* query  = (const float*)d_in[0];   // [B, D]
    const float* values = (const float*)d_in[1];   // [B, S, D]
    const float* W1     = (const float*)d_in[2];   // [D, U]
    const float* b1     = (const float*)d_in[3];   // [U]
    const float* W2     = (const float*)d_in[4];   // [D, U]
    const float* b2     = (const float*)d_in[5];   // [U]
    const float* Vvec   = (const float*)d_in[6];   // [U, 1]
    const float* bV     = (const float*)d_in[7];   // [1]

    // Outputs: context [B,D] then attn [B,S,1]
    float* ctx  = (float*)d_out;
    float* attn = (float*)d_out + (size_t)B_DIM * D_DIM;

    // Workspace layout (256B-aligned segments), ~2.8 MB total
    char* ws = (char*)d_ws;
    unsigned short* w1t = (unsigned short*)(ws);                        // 256 KB
    float* pqc     = (float*)(ws + 262144);                             // 64 KB
    float* score   = (float*)(ws + 262144 + 65536);                     // 512 KB
    float* partial = (float*)(ws + 262144 + 65536 + 524288);            // 2 MB

    w1t_kernel        <<<D_DIM, 256, 0, stream>>>(W1, w1t);
    pq_kernel         <<<B_DIM, 256, 0, stream>>>(query, W2, b2, b1, pqc);
    score_kernel      <<<B_DIM * (S_DIM / SROWS), 512, 0, stream>>>(values, w1t, pqc, Vvec, bV, score);
    softmax_kernel    <<<B_DIM, 256, 0, stream>>>(score, attn);
    ctx_partial_kernel<<<B_DIM * NS, 256, 0, stream>>>(values, attn, partial);
    ctx_reduce_kernel <<<B_DIM, 512, 0, stream>>>(partial, ctx);
}
